// TransformerD_22393959481803
// MI455X (gfx1250) — matrix operand order repaired and checked by reading, not compiled, not run on hardware
//
#include <hip/hip_runtime.h>

// ---------------------------------------------------------------------------
// CDNA5 (gfx1250) implementation: bf16 WMMA (V_WMMA_F32_16X16X32_BF16) for the
// QKV projections and grouped local attention; fp32 VALU for softmax / ReLU /
// LayerNorm.  Wave32; operand layouts per CDNA5 ISA §7.12.2.
//
// Round-2 changes (driven by round-1 disassembly):
//  * W pre-transposed to bf16 [N][K] so GEMM B-operands are single contiguous
//    32B vector loads (was 16 scalar global_load_b32 + cvt per K-step).
//  * GEMM waves hold 3 accumulators and reuse each A fragment for 3 WMMAs.
//  * Attention computed transposed (S^T = K*Q^T, out^T = V^T*P^T): softmax is
//    in-register + one shfl_xor(16); P^T fed to WMMA straight from score
//    registers via 8 shuffles per K-step (probs-LDS round trip eliminated);
//    masks load as 8B words; outputs stored as one 32B vector per lane.
// ---------------------------------------------------------------------------

typedef __attribute__((ext_vector_type(16))) __bf16 v16bf;
typedef __attribute__((ext_vector_type(8)))  __bf16 bf16x8;
typedef __attribute__((ext_vector_type(4)))  __bf16 bf16x4;
typedef __attribute__((ext_vector_type(2)))  __bf16 bf16x2;
typedef __attribute__((ext_vector_type(8)))  float  v8f;
typedef __attribute__((ext_vector_type(4)))  float  f32x4;

constexpr int SLc = 128;   // local window
constexpr int Dc  = 384;   // model dim
constexpr int Hc  = 48;    // heads
constexpr int Ec  = 8;     // head dim
constexpr int Gc  = 30;    // groups per batch
constexpr int NBc = 240;   // B*G
constexpr int NRc = 30720; // NBc*SLc rows

static __device__ __forceinline__ v8f wmma_bf16f32(v16bf a, v16bf b, v8f c) {
  // (neg_a, A, neg_b, B, c_mod, C, reuse_a, reuse_b)
  return __builtin_amdgcn_wmma_f32_16x16x32_bf16(false, a, false, b, (short)0, c,
                                                 false, false);
}

static __device__ __forceinline__ v8f zero_v8f() {
  v8f z = {0.f, 0.f, 0.f, 0.f, 0.f, 0.f, 0.f, 0.f};
  return z;
}

// ---------------------------------------------------------------------------
// Kernel 0: pack W (fp32, [K][N]) -> W^T (bf16, [N][K]).  Makes every GEMM
// B-operand a contiguous 32B run.  Total 3 x 576KB read, 3 x 288KB written.
// ---------------------------------------------------------------------------
__global__ __launch_bounds__(256) void pack_wT(
    const float* __restrict__ Wq, const float* __restrict__ Wk,
    const float* __restrict__ Wv,
    __bf16* __restrict__ Tq, __bf16* __restrict__ Tk, __bf16* __restrict__ Tv)
{
  const int w = blockIdx.y;
  const float* W = (w == 0) ? Wq : (w == 1) ? Wk : Wv;
  __bf16*      T = (w == 0) ? Tq : (w == 1) ? Tk : Tv;
  const int idx = blockIdx.x * 256 + threadIdx.x;  // 0..147455 = n*384 + k
  const int n = idx / Dc, k = idx - n * Dc;
  T[idx] = (__bf16)W[(size_t)k * Dc + n];
}

// ---------------------------------------------------------------------------
// Kernel 1: QKV projection GEMM.  C[30720x384] = X * W.  Block = 8 waves owns
// one 16-row strip (LDS, bf16); each wave owns 3 of the 24 N-tiles and reuses
// the A fragment for 3 WMMAs per K-step.  blockIdx.y selects Q/K/V.
// ---------------------------------------------------------------------------
__global__ __launch_bounds__(256) void qkv_gemm(
    const float* __restrict__ X, const __bf16* __restrict__ WTq,
    const __bf16* __restrict__ WTk, const __bf16* __restrict__ WTv,
    __bf16* __restrict__ Q, __bf16* __restrict__ K, __bf16* __restrict__ V)
{
  __shared__ __bf16 lds_a[16 * Dc];

  const int strip = blockIdx.x;       // 0..1919
  const int wsel  = blockIdx.y;       // 0..2
  const __bf16* WT = (wsel == 0) ? WTq : (wsel == 1) ? WTk : WTv;
  __bf16*       Om = (wsel == 0) ? Q   : (wsel == 1) ? K   : V;

  // Stage A strip (16x384) fp32 -> bf16, vectorized (b128 in, b64 out).
  for (int i4 = threadIdx.x; i4 < 16 * Dc / 4; i4 += 256) {
    f32x4 x = *(const f32x4*)&X[(size_t)strip * (16 * Dc) + i4 * 4];
    bf16x4 y;
#pragma unroll
    for (int j = 0; j < 4; ++j) y[j] = (__bf16)x[j];
    *(bf16x4*)&lds_a[i4 * 4] = y;
  }
  __syncthreads();

  const int wave = threadIdx.x >> 5;
  const int lane = threadIdx.x & 31;
  const int half = lane >> 4;
  const int l15  = lane & 15;
  const int col0 = (wave + 0) * 16 + l15;
  const int col1 = (wave + 8) * 16 + l15;
  const int col2 = (wave + 16) * 16 + l15;

  v8f acc0 = zero_v8f(), acc1 = zero_v8f(), acc2 = zero_v8f();
#pragma unroll
  for (int kk = 0; kk < Dc / 32; ++kk) {
    // A fragment: lane half selects contiguous K run kb..kb+15.
    const int kb = kk * 32 + half * 16;
    bf16x8 lo = *(const bf16x8*)&lds_a[l15 * Dc + kb];
    bf16x8 hi = *(const bf16x8*)&lds_a[l15 * Dc + kb + 8];
    v16bf a = __builtin_shufflevector(lo, hi, 0, 1, 2, 3, 4, 5, 6, 7, 8, 9, 10,
                                      11, 12, 13, 14, 15);
    // B fragments: contiguous 32B from W^T.
    const int kr0 = kk * 32 + half * 16;
    v16bf b0 = *(const v16bf*)&WT[(size_t)col0 * Dc + kr0];
    v16bf b1 = *(const v16bf*)&WT[(size_t)col1 * Dc + kr0];
    v16bf b2 = *(const v16bf*)&WT[(size_t)col2 * Dc + kr0];
    acc0 = wmma_bf16f32(a, b0, acc0);
    acc1 = wmma_bf16f32(a, b1, acc1);
    acc2 = wmma_bf16f32(a, b2, acc2);
  }

#pragma unroll
  for (int r = 0; r < 8; ++r) {
    const size_t row = (size_t)strip * 16 + r + half * 8;
    Om[row * Dc + col0] = (__bf16)acc0[r];
    Om[row * Dc + col1] = (__bf16)acc1[r];
    Om[row * Dc + col2] = (__bf16)acc2[r];
  }
}

// ---------------------------------------------------------------------------
// Kernel 2: conquer-token query projection  qc[b,:] = conquer[b,:] @ Wq.
// ---------------------------------------------------------------------------
__global__ __launch_bounds__(384) void qc_proj(
    const float* __restrict__ cq, const float* __restrict__ Wq,
    float* __restrict__ qc)
{
  const int b = blockIdx.x, col = threadIdx.x;
  float s = 0.f;
  for (int kk = 0; kk < Dc; ++kk)
    s += cq[b * Dc + kk] * Wq[kk * Dc + col];
  qc[b * Dc + col] = s;
}

// ---------------------------------------------------------------------------
// Kernel 3: grouped local attention, transposed formulation.
// One wave per (n, head, 16-row M tile):
//   S^T tiles (t x m) = K(16x8 pad32) * Q^T(8x16 pad)   -> 8 WMMAs
//   each lane owns row m = mtile*16+l15 with 8 consecutive t per register
//   block -> 8B mask loads, in-register softmax + one shfl_xor(16)
//   out^T (e x m) = V^T(8x128 pad16) * P^T(128x16)       -> 4 WMMAs
//   P^T built from score registers (8 shuffles / K-step), V^T staged in LDS.
// Raw result written to d_out; kernel 4 applies ReLU+LN in place.
// ---------------------------------------------------------------------------
__global__ __launch_bounds__(32) void attn_local(
    const __bf16* __restrict__ Q, const __bf16* __restrict__ K,
    const __bf16* __restrict__ V, const unsigned char* __restrict__ mask,
    float* __restrict__ out)
{
  __shared__ __bf16 lds_vt[Ec * SLc];  // V^T: [e][t]

  const int idx   = blockIdx.x;
  const int mtile = idx & 7;
  const int h     = (idx >> 3) % Hc;
  const int n     = idx / (Hc * 8);
  const int lane  = threadIdx.x;
  const int half  = lane >> 4;
  const int l15   = lane & 15;

  // Stage V^T: lane loads rows t0,t0+1 and stores packed bf16 pairs.
#pragma unroll
  for (int rp = 0; rp < 2; ++rp) {
    const int t0 = lane * 4 + rp * 2;
    bf16x8 r0 = *(const bf16x8*)&V[((size_t)n * SLc + t0) * Dc + h * Ec];
    bf16x8 r1 = *(const bf16x8*)&V[((size_t)n * SLc + t0 + 1) * Dc + h * Ec];
#pragma unroll
    for (int e = 0; e < 8; ++e) {
      bf16x2 p;
      p[0] = r0[e];
      p[1] = r1[e];
      *(bf16x2*)&lds_vt[e * SLc + t0] = p;
    }
  }

  // Q^T as B operand (K = E zero-padded to 32): lanes<16, elements 0..7 real.
  v16bf bq;
#pragma unroll
  for (int e = 0; e < 16; ++e) bq[e] = (__bf16)0.f;
  if (lane < 16) {
    bf16x8 qrow =
        *(const bf16x8*)&Q[((size_t)n * SLc + mtile * 16 + lane) * Dc + h * Ec];
#pragma unroll
    for (int e = 0; e < 8; ++e) bq[e] = qrow[e];
  }

  // S^T tiles: ct[tt][r] = score(m = mtile*16+l15, t = tt*16 + half*8 + r).
  v8f ct[8];
#pragma unroll
  for (int tt = 0; tt < 8; ++tt) {
    v16bf ak;
#pragma unroll
    for (int e = 0; e < 16; ++e) ak[e] = (__bf16)0.f;
    if (lane < 16) {
      bf16x8 krow =
          *(const bf16x8*)&K[((size_t)n * SLc + tt * 16 + lane) * Dc + h * Ec];
#pragma unroll
      for (int e = 0; e < 8; ++e) ak[e] = krow[e];
    }
    ct[tt] = wmma_bf16f32(ak, bq, zero_v8f());
  }

  // Mask (8 consecutive bytes per tile) + softmax over t for this lane's row.
  const float NEG = -__builtin_inff();
  const unsigned char* mrow =
      mask + ((size_t)n * SLc + mtile * 16 + l15) * SLc;
  float mx = NEG;
#pragma unroll
  for (int tt = 0; tt < 8; ++tt) {
    const uint2 mv = *(const uint2*)(mrow + tt * 16 + half * 8);
#pragma unroll
    for (int r = 0; r < 8; ++r) {
      const unsigned byte = ((r < 4 ? mv.x : mv.y) >> (8 * (r & 3))) & 0xffu;
      if (byte) ct[tt][r] = NEG;
      mx = fmaxf(mx, ct[tt][r]);
    }
  }
  mx = fmaxf(mx, __shfl_xor(mx, 16, 32));  // other half holds other t's
  float ssum = 0.f;
#pragma unroll
  for (int tt = 0; tt < 8; ++tt)
#pragma unroll
    for (int r = 0; r < 8; ++r) {
      const float p = __expf(ct[tt][r] - mx);
      ct[tt][r] = p;
      ssum += p;
    }
  ssum += __shfl_xor(ssum, 16, 32);
  const float rinv = 1.f / ssum;
#pragma unroll
  for (int tt = 0; tt < 8; ++tt)
#pragma unroll
    for (int r = 0; r < 8; ++r) ct[tt][r] *= rinv;

  __syncthreads();  // V^T staging visible

  // out^T = V^T * P^T, K loop over t in 4 x 32.
  v8f acc = zero_v8f();
#pragma unroll
  for (int kk = 0; kk < 4; ++kk) {
    // A: V^T rows e (rows 8..15 padding -> lanes with l15>=8 are zero).
    const int kb = kk * 32 + half * 16;
    v16bf av;
#pragma unroll
    for (int e = 0; e < 16; ++e) av[e] = (__bf16)0.f;
    if (l15 < Ec) {
      bf16x8 lo = *(const bf16x8*)&lds_vt[l15 * SLc + kb];
      bf16x8 hi = *(const bf16x8*)&lds_vt[l15 * SLc + kb + 8];
      av = __builtin_shufflevector(lo, hi, 0, 1, 2, 3, 4, 5, 6, 7, 8, 9, 10, 11,
                                   12, 13, 14, 15);
    }
    // B: P^T column m = l15; swap the half-tiles between lane halves.
    float recv[8];
#pragma unroll
    for (int e = 0; e < 8; ++e) {
      const float send = half ? ct[2 * kk][e] : ct[2 * kk + 1][e];
      recv[e] = __shfl_xor(send, 16, 32);
    }
    v16bf bp;
#pragma unroll
    for (int e = 0; e < 8; ++e) {
      const float own = half ? ct[2 * kk + 1][e] : ct[2 * kk][e];
      bp[e]     = (__bf16)(half ? recv[e] : own);
      bp[8 + e] = (__bf16)(half ? own : recv[e]);
    }
    acc = wmma_bf16f32(av, bp, acc);
  }

  // D rows = e (0..7 real, lanes<16), cols = m: 8 consecutive floats per lane.
  if (lane < 16) {
    float* op = &out[((size_t)n * SLc + mtile * 16 + lane) * Dc + h * Ec];
    *(v8f*)op = acc;  // 32B-aligned vector store
  }
}

// ---------------------------------------------------------------------------
// Kernel 4: in-place ReLU + LayerNorm over each 384-wide row of d_out.
// One wave per row, float4 vector path, butterfly reductions.
// ---------------------------------------------------------------------------
__global__ __launch_bounds__(256) void relu_ln(
    float* __restrict__ out, const float* __restrict__ gamma,
    const float* __restrict__ beta)
{
  const int row  = blockIdx.x * 8 + (threadIdx.x >> 5);
  const int lane = threadIdx.x & 31;
  f32x4 v[3];
  float s = 0.f;
#pragma unroll
  for (int i = 0; i < 3; ++i) {
    f32x4 x = *(const f32x4*)&out[(size_t)row * Dc + i * 128 + lane * 4];
#pragma unroll
    for (int j = 0; j < 4; ++j) {
      x[j] = fmaxf(x[j], 0.f);
      s += x[j];
    }
    v[i] = x;
  }
#pragma unroll
  for (int off = 16; off >= 1; off >>= 1) s += __shfl_xor(s, off, 32);
  const float mu = s * (1.f / Dc);
  float q = 0.f;
#pragma unroll
  for (int i = 0; i < 3; ++i)
#pragma unroll
    for (int j = 0; j < 4; ++j) {
      const float d = v[i][j] - mu;
      q += d * d;
    }
#pragma unroll
  for (int off = 16; off >= 1; off >>= 1) q += __shfl_xor(q, off, 32);
  const float inv = rsqrtf(q * (1.f / Dc) + 1e-5f);
#pragma unroll
  for (int i = 0; i < 3; ++i) {
    const int c = i * 128 + lane * 4;
    const f32x4 gg = *(const f32x4*)&gamma[c];
    const f32x4 bb = *(const f32x4*)&beta[c];
    f32x4 o;
#pragma unroll
    for (int j = 0; j < 4; ++j) o[j] = (v[i][j] - mu) * inv * gg[j] + bb[j];
    *(f32x4*)&out[(size_t)row * Dc + c] = o;
  }
}

// ---------------------------------------------------------------------------
// Kernel 5: conquer-token cross attention. One wave per (n, h); lane owns 4 of
// the 128 keys.  Single query row -> scalar math; vector b128 K/V reads.
// ---------------------------------------------------------------------------
__global__ __launch_bounds__(32) void attn_conquer(
    const float* __restrict__ qc, const __bf16* __restrict__ K,
    const __bf16* __restrict__ V, const unsigned char* __restrict__ maskc,
    float* __restrict__ out2)
{
  const int idx  = blockIdx.x;  // n*H + h
  const int h    = idx % Hc;
  const int n    = idx / Hc;
  const int b    = n / Gc;
  const int lane = threadIdx.x;

  float qv[8];
#pragma unroll
  for (int e = 0; e < 8; ++e) qv[e] = qc[b * Dc + h * Ec + e];

  const float NEG = -__builtin_inff();
  float sc[4];
#pragma unroll
  for (int j = 0; j < 4; ++j) {
    const int t = j * 32 + lane;
    const bf16x8 kv = *(const bf16x8*)&K[((size_t)n * SLc + t) * Dc + h * Ec];
    float s = 0.f;
#pragma unroll
    for (int e = 0; e < 8; ++e) s += qv[e] * (float)kv[e];
    if (maskc[(size_t)n * SLc + t]) s = NEG;
    sc[j] = s;
  }
  float m = fmaxf(fmaxf(sc[0], sc[1]), fmaxf(sc[2], sc[3]));
#pragma unroll
  for (int off = 16; off >= 1; off >>= 1) m = fmaxf(m, __shfl_xor(m, off, 32));

  float a[8];
#pragma unroll
  for (int e = 0; e < 8; ++e) a[e] = 0.f;
  float ssum = 0.f;
#pragma unroll
  for (int j = 0; j < 4; ++j) {
    const int t = j * 32 + lane;
    const float p = __expf(sc[j] - m);
    ssum += p;
    const bf16x8 vv = *(const bf16x8*)&V[((size_t)n * SLc + t) * Dc + h * Ec];
#pragma unroll
    for (int e = 0; e < 8; ++e) a[e] += p * (float)vv[e];
  }
#pragma unroll
  for (int off = 16; off >= 1; off >>= 1) ssum += __shfl_xor(ssum, off, 32);
#pragma unroll
  for (int e = 0; e < 8; ++e)
#pragma unroll
    for (int off = 16; off >= 1; off >>= 1) a[e] += __shfl_xor(a[e], off, 32);

  if (lane == 0) {
    const float inv = 1.f / ssum;
#pragma unroll
    for (int e = 0; e < 8; ++e)
      out2[(size_t)n * Dc + h * Ec + e] = fmaxf(a[e] * inv, 0.f);
  }
}

// ---------------------------------------------------------------------------
extern "C" void kernel_launch(void* const* d_in, const int* in_sizes, int n_in,
                              void* d_out, int out_size, void* d_ws,
                              size_t ws_size, hipStream_t stream)
{
  (void)in_sizes; (void)n_in; (void)out_size; (void)ws_size;

  const float*         x     = (const float*)d_in[0];          // [B,S,D]
  const float*         cq    = (const float*)d_in[1];          // [B,1,D]
  const unsigned char* ml    = (const unsigned char*)d_in[2];  // [B,G,SL,SL]
  const unsigned char* mc    = (const unsigned char*)d_in[3];  // [B,G,SL]
  const float*         Wq    = (const float*)d_in[4];
  const float*         Wk    = (const float*)d_in[5];
  const float*         Wv    = (const float*)d_in[6];
  const float*         gamma = (const float*)d_in[7];
  const float*         beta  = (const float*)d_in[8];

  char* ws = (char*)d_ws;
  const size_t qkv_bytes = (size_t)NRc * Dc * sizeof(__bf16);  // 23.6 MB each
  __bf16* Qb  = (__bf16*)(ws);
  __bf16* Kb  = (__bf16*)(ws + qkv_bytes);
  __bf16* Vb  = (__bf16*)(ws + 2 * qkv_bytes);
  float*  qcb = (float*)(ws + 3 * qkv_bytes);                  // [8, 384]
  const size_t wt_off = 3 * qkv_bytes + 16384;
  const size_t wt_bytes = (size_t)Dc * Dc * sizeof(__bf16);    // 288 KB each
  __bf16* WTq = (__bf16*)(ws + wt_off);
  __bf16* WTk = (__bf16*)(ws + wt_off + wt_bytes);
  __bf16* WTv = (__bf16*)(ws + wt_off + 2 * wt_bytes);

  float* out1 = (float*)d_out;            // [B,S,D] raw -> LN'd in place
  float* out2 = out1 + (size_t)NRc * Dc;  // [B,G,D]

  pack_wT<<<dim3(Dc * Dc / 256, 3), 256, 0, stream>>>(Wq, Wk, Wv, WTq, WTk, WTv);
  qkv_gemm<<<dim3(NRc / 16, 3), 256, 0, stream>>>(x, WTq, WTk, WTv, Qb, Kb, Vb);
  qc_proj<<<dim3(8), 384, 0, stream>>>(cq, Wq, qcb);
  attn_local<<<dim3(NBc * Hc * 8), 32, 0, stream>>>(Qb, Kb, Vb, ml, out1);
  relu_ln<<<dim3(NRc / 8), 256, 0, stream>>>(out1, gamma, beta);
  attn_conquer<<<dim3(NBc * Hc), 32, 0, stream>>>(qcb, Kb, Vb, mc, out2);
}
